// GAT_32512902431420
// MI455X (gfx1250) — matrix operand order repaired and checked by reading, not compiled, not run on hardware
//
#include <hip/hip_runtime.h>
#include <hip/hip_bf16.h>

#define NNODES 50000
#define NEDGES 800000
#define ETOT   (NEDGES + NNODES)
#define FIN    500
#define KPAD   512      // K padded to a multiple of 32 -> no tail, 16B-aligned rows
#define NHEAD  8
#define CH     16
#define HC     128      // NHEAD * CH
#define NCLS   7
#define NEGSL  0.2f

typedef __attribute__((ext_vector_type(16))) _Float16 v16h;  // 32B
typedef __attribute__((ext_vector_type(8)))  _Float16 v8h;   // 16B
typedef __attribute__((ext_vector_type(8)))  float    v8f;

// ---------------- utility ----------------

__global__ void fill_kernel(float* __restrict__ p, int n, float v) {
    int i = blockIdx.x * blockDim.x + threadIdx.x;
    if (i < n) p[i] = v;
}

// convert X [N, FIN] f32 -> Xh [N, KPAD] f16, zero-padded columns
__global__ void cvtX_kernel(const float* __restrict__ in, _Float16* __restrict__ out) {
    int idx = blockIdx.x * blockDim.x + threadIdx.x;
    if (idx >= NNODES * KPAD) return;
    int row = idx >> 9;          // / KPAD
    int col = idx & (KPAD - 1);
    out[idx] = (col < FIN) ? (_Float16)in[(size_t)row * FIN + col] : (_Float16)0.f;
}

// convert W1 [FIN, HC] f32 -> W1h [KPAD, HC] f16, zero-padded rows
__global__ void cvtW_kernel(const float* __restrict__ in, _Float16* __restrict__ out) {
    int idx = blockIdx.x * blockDim.x + threadIdx.x;
    if (idx >= KPAD * HC) return;
    int k = idx >> 7;            // / HC
    out[idx] = (k < FIN) ? (_Float16)in[idx] : (_Float16)0.f;
}

__device__ __forceinline__ void edge_get(const int* __restrict__ ei, int eid,
                                         int& s, int& d) {
    if (eid < NEDGES) { s = ei[eid]; d = ei[NEDGES + eid]; }
    else              { s = eid - NEDGES; d = s; }   // self loops
}

__device__ __forceinline__ void atomicMaxF(float* addr, float val) {
    int* ai = (int*)addr;
    int cur = __float_as_int(*addr);
    while (__int_as_float(cur) < val) {
        int old = atomicCAS(ai, cur, __float_as_int(val));
        if (old == cur) break;
        cur = old;
    }
}

__device__ __forceinline__ float lrelu(float x) {
    return x > 0.f ? x : NEGSL * x;
}

// ---------------- layer 1 GEMM: hfeat = Xh @ W1h (f16 WMMA, f32 acc) ----------------
// One wave computes a 16(M) x 128(N) tile. 50000 = 3125 * 16 exactly -> no M tail.
// KPAD = 512 = 16 * 32 -> no K tail, all fragments naturally aligned.

__global__ __launch_bounds__(128)
void gemm1_wmma(const _Float16* __restrict__ Xh, const _Float16* __restrict__ W1h,
                float* __restrict__ hfeat) {
    const int lane = threadIdx.x & 31;
    const int wave = threadIdx.x >> 5;
    const int tile = blockIdx.x * 4 + wave;          // 16-row tile index
    if (tile * 16 >= NNODES) return;                 // wave-uniform: EXEC stays all-1
    const int mlo = lane & 15;
    const int hi  = lane >> 4;
    const _Float16* Arow = Xh + (size_t)(tile * 16 + mlo) * KPAD;

    v8f acc[8];
#pragma unroll
    for (int t = 0; t < 8; ++t)
#pragma unroll
        for (int i = 0; i < 8; ++i) acc[t][i] = 0.f;

#pragma unroll 2
    for (int k0 = 0; k0 < KPAD; k0 += 32) {
        // A fragment: lane holds row m, elements K = (e>>3)*16 + hi*8 + (e&7)
        const _Float16* ap = Arow + k0 + hi * 8;
        v8h alo = *(const v8h*)(ap);          // K = k0 + hi*8 + 0..7
        v8h ahi = *(const v8h*)(ap + 16);     // K = k0 + 16 + hi*8 + 0..7
        v16h a = __builtin_shufflevector(alo, ahi,
                 0, 1, 2, 3, 4, 5, 6, 7, 8, 9, 10, 11, 12, 13, 14, 15);
        __builtin_prefetch(ap + 64, 0, 3);    // gfx1250 global_prefetch_b8

        // B fragments: lane holds column n = t*16 + mlo, elements K = (e>>3)*16 + hi*8 + (e&7)
        const _Float16* bp = W1h + (size_t)k0 * HC + mlo;
#pragma unroll
        for (int t = 0; t < 8; ++t) {
            v16h b;
#pragma unroll
            for (int s = 0; s < 16; ++s)
                b[s] = bp[(size_t)((s >> 3) * 16 + hi * 8 + (s & 7)) * HC + t * 16];
            acc[t] = __builtin_amdgcn_wmma_f32_16x16x32_f16(
                false, a, false, b, (short)0, acc[t], false, false);
        }
    }

    // store D: acc[t][r] -> row tile*16 + hi*8 + r, col t*16 + mlo
    const int orow = tile * 16 + hi * 8;
#pragma unroll
    for (int t = 0; t < 8; ++t)
#pragma unroll
        for (int r = 0; r < 8; ++r)
            hfeat[(size_t)(orow + r) * HC + t * 16 + mlo] = acc[t][r];
}

// ---------------- layer 1 attention coefficients ----------------

__global__ void a1coef_kernel(const float* __restrict__ hfeat,
                              const float* __restrict__ att_s,
                              const float* __restrict__ att_d,
                              float* __restrict__ asrc, float* __restrict__ adst) {
    int idx = blockIdx.x * blockDim.x + threadIdx.x;   // n*8 + h
    if (idx >= NNODES * NHEAD) return;
    int n = idx >> 3, h = idx & 7;
    const float* hp = hfeat + (size_t)n * HC + h * CH;
    const float* sp = att_s + h * CH;
    const float* dp = att_d + h * CH;
    float s = 0.f, d = 0.f;
#pragma unroll
    for (int c = 0; c < CH; ++c) { s += hp[c] * sp[c]; d += hp[c] * dp[c]; }
    asrc[idx] = s;
    adst[idx] = d;
}

// ---------------- layer 1 edge passes (segment softmax + message) ----------------

__global__ void edge_max1(const int* __restrict__ ei,
                          const float* __restrict__ asrc, const float* __restrict__ adst,
                          float* __restrict__ emax) {
    int idx = blockIdx.x * blockDim.x + threadIdx.x;
    if (idx >= ETOT * NHEAD) return;
    int eid = idx >> 3, h = idx & 7, s, d;
    edge_get(ei, eid, s, d);
    float e = lrelu(asrc[s * NHEAD + h] + adst[d * NHEAD + h]);
    atomicMaxF(&emax[d * NHEAD + h], e);
}

__global__ void edge_sum1(const int* __restrict__ ei,
                          const float* __restrict__ asrc, const float* __restrict__ adst,
                          const float* __restrict__ emax, float* __restrict__ esum) {
    int idx = blockIdx.x * blockDim.x + threadIdx.x;
    if (idx >= ETOT * NHEAD) return;
    int eid = idx >> 3, h = idx & 7, s, d;
    edge_get(ei, eid, s, d);
    float e = lrelu(asrc[s * NHEAD + h] + adst[d * NHEAD + h]);
    atomicAdd(&esum[d * NHEAD + h], __expf(e - emax[d * NHEAD + h]));
}

__global__ void edge_msg1(const int* __restrict__ ei,
                          const float* __restrict__ asrc, const float* __restrict__ adst,
                          const float* __restrict__ emax, const float* __restrict__ esum,
                          const float* __restrict__ hfeat, float* __restrict__ acc1) {
    int idx = blockIdx.x * blockDim.x + threadIdx.x;
    if (idx >= ETOT * NHEAD) return;
    int eid = idx >> 3, h = idx & 7, s, d;
    edge_get(ei, eid, s, d);
    float e = lrelu(asrc[s * NHEAD + h] + adst[d * NHEAD + h]);
    float alpha = __expf(e - emax[d * NHEAD + h]) / (esum[d * NHEAD + h] + 1e-16f);
    const float* hs = hfeat + (size_t)s * HC + h * CH;
    float* od = acc1 + (size_t)d * HC + h * CH;
#pragma unroll
    for (int c = 0; c < CH; ++c) atomicAdd(&od[c], hs[c] * alpha);
}

// ---------------- node kernel: bias+ReLU, layer-2 GEMM (128->7), attn dots ----------------

__global__ void node2_kernel(const float* __restrict__ acc1, const float* __restrict__ b1,
                             const float* __restrict__ W2,
                             const float* __restrict__ att_s2, const float* __restrict__ att_d2,
                             float* __restrict__ h2, float* __restrict__ asrc2,
                             float* __restrict__ adst2) {
    int n = blockIdx.x * blockDim.x + threadIdx.x;
    if (n >= NNODES) return;
    const float* row = acc1 + (size_t)n * HC;
    float o[NCLS];
#pragma unroll
    for (int c = 0; c < NCLS; ++c) o[c] = 0.f;
    for (int k = 0; k < HC; ++k) {
        float v = row[k] + b1[k];
        v = v > 0.f ? v : 0.f;                    // ReLU between layers
        const float* w = W2 + k * NCLS;
#pragma unroll
        for (int c = 0; c < NCLS; ++c) o[c] += v * w[c];
    }
    float s = 0.f, d = 0.f;
#pragma unroll
    for (int c = 0; c < NCLS; ++c) {
        h2[n * NCLS + c] = o[c];
        s += o[c] * att_s2[c];
        d += o[c] * att_d2[c];
    }
    asrc2[n] = s;
    adst2[n] = d;
}

// ---------------- layer 2 edge passes (H=1, C=7) ----------------

__global__ void edge_max2(const int* __restrict__ ei,
                          const float* __restrict__ asrc, const float* __restrict__ adst,
                          float* __restrict__ emax) {
    int eid = blockIdx.x * blockDim.x + threadIdx.x;
    if (eid >= ETOT) return;
    int s, d; edge_get(ei, eid, s, d);
    atomicMaxF(&emax[d], lrelu(asrc[s] + adst[d]));
}

__global__ void edge_sum2(const int* __restrict__ ei,
                          const float* __restrict__ asrc, const float* __restrict__ adst,
                          const float* __restrict__ emax, float* __restrict__ esum) {
    int eid = blockIdx.x * blockDim.x + threadIdx.x;
    if (eid >= ETOT) return;
    int s, d; edge_get(ei, eid, s, d);
    float e = lrelu(asrc[s] + adst[d]);
    atomicAdd(&esum[d], __expf(e - emax[d]));
}

__global__ void edge_msg2(const int* __restrict__ ei,
                          const float* __restrict__ asrc, const float* __restrict__ adst,
                          const float* __restrict__ emax, const float* __restrict__ esum,
                          const float* __restrict__ h2, float* __restrict__ acc2) {
    int eid = blockIdx.x * blockDim.x + threadIdx.x;
    if (eid >= ETOT) return;
    int s, d; edge_get(ei, eid, s, d);
    float e = lrelu(asrc[s] + adst[d]);
    float alpha = __expf(e - emax[d]) / (esum[d] + 1e-16f);
    const float* hs = h2 + (size_t)s * NCLS;
    float* od = acc2 + (size_t)d * NCLS;
#pragma unroll
    for (int c = 0; c < NCLS; ++c) atomicAdd(&od[c], hs[c] * alpha);
}

__global__ void finalize_kernel(const float* __restrict__ acc2,
                                const float* __restrict__ b2, float* __restrict__ out) {
    int idx = blockIdx.x * blockDim.x + threadIdx.x;
    if (idx >= NNODES * NCLS) return;
    out[idx] = acc2[idx] + b2[idx % NCLS];
}

// ---------------- host launch ----------------

extern "C" void kernel_launch(void* const* d_in, const int* in_sizes, int n_in,
                              void* d_out, int out_size, void* d_ws, size_t ws_size,
                              hipStream_t stream) {
    const float* x        = (const float*)d_in[0];
    const int*   ei       = (const int*)  d_in[1];   // [2, E] flattened
    const float* W1       = (const float*)d_in[2];
    const float* att_src1 = (const float*)d_in[3];
    const float* att_dst1 = (const float*)d_in[4];
    const float* b1       = (const float*)d_in[5];
    const float* W2       = (const float*)d_in[6];
    const float* att_src2 = (const float*)d_in[7];
    const float* att_dst2 = (const float*)d_in[8];
    const float* b2       = (const float*)d_in[9];
    float* out = (float*)d_out;

    // workspace layout (floats, then f16 region; all offsets stay 32B aligned)
    float* ws    = (float*)d_ws;
    float* hfeat = ws;                          // 128*N
    float* acc1  = hfeat + (size_t)HC * NNODES; // 128*N
    float* asrc1 = acc1  + (size_t)HC * NNODES; // 8*N
    float* adst1 = asrc1 + (size_t)NHEAD * NNODES;
    float* emax1 = adst1 + (size_t)NHEAD * NNODES;
    float* esum1 = emax1 + (size_t)NHEAD * NNODES;
    float* h2    = esum1 + (size_t)NHEAD * NNODES; // 7*N
    float* asrc2 = h2    + (size_t)NCLS * NNODES;  // N
    float* adst2 = asrc2 + NNODES;
    float* emax2 = adst2 + NNODES;
    float* esum2 = emax2 + NNODES;
    float* acc2  = esum2 + NNODES;                 // 7*N
    _Float16* xh  = (_Float16*)(acc2 + (size_t)NCLS * NNODES); // N*KPAD f16
    _Float16* w1h = xh + (size_t)NNODES * KPAD;                // KPAD*HC f16

    const int TB = 256;
    auto blocks = [](long long n, int tb) { return (int)((n + tb - 1) / tb); };

    // init accumulators / softmax state
    fill_kernel<<<blocks(NHEAD * NNODES, TB), TB, 0, stream>>>(emax1, NHEAD * NNODES, -1e30f);
    fill_kernel<<<blocks(NHEAD * NNODES, TB), TB, 0, stream>>>(esum1, NHEAD * NNODES, 0.f);
    fill_kernel<<<blocks((long long)HC * NNODES, TB), TB, 0, stream>>>(acc1, HC * NNODES, 0.f);
    fill_kernel<<<blocks(NNODES, TB), TB, 0, stream>>>(emax2, NNODES, -1e30f);
    fill_kernel<<<blocks(NNODES, TB), TB, 0, stream>>>(esum2, NNODES, 0.f);
    fill_kernel<<<blocks(NCLS * NNODES, TB), TB, 0, stream>>>(acc2, NCLS * NNODES, 0.f);

    // precision conversion for the WMMA GEMM
    cvtX_kernel<<<blocks((long long)NNODES * KPAD, TB), TB, 0, stream>>>(x, xh);
    cvtW_kernel<<<blocks(KPAD * HC, TB), TB, 0, stream>>>(W1, w1h);

    // layer 1
    {
        const int ntiles = NNODES / 16;                  // 3125
        gemm1_wmma<<<(ntiles + 3) / 4, 128, 0, stream>>>(xh, w1h, hfeat);
    }
    a1coef_kernel<<<blocks(NNODES * NHEAD, TB), TB, 0, stream>>>(hfeat, att_src1, att_dst1, asrc1, adst1);
    edge_max1<<<blocks((long long)ETOT * NHEAD, TB), TB, 0, stream>>>(ei, asrc1, adst1, emax1);
    edge_sum1<<<blocks((long long)ETOT * NHEAD, TB), TB, 0, stream>>>(ei, asrc1, adst1, emax1, esum1);
    edge_msg1<<<blocks((long long)ETOT * NHEAD, TB), TB, 0, stream>>>(ei, asrc1, adst1, emax1, esum1, hfeat, acc1);

    // between layers + layer 2 linear/attention coefficients
    node2_kernel<<<blocks(NNODES, TB), TB, 0, stream>>>(acc1, b1, W2, att_src2, att_dst2, h2, asrc2, adst2);

    // layer 2 softmax aggregation
    edge_max2<<<blocks(ETOT, TB), TB, 0, stream>>>(ei, asrc2, adst2, emax2);
    edge_sum2<<<blocks(ETOT, TB), TB, 0, stream>>>(ei, asrc2, adst2, emax2, esum2);
    edge_msg2<<<blocks(ETOT, TB), TB, 0, stream>>>(ei, asrc2, adst2, emax2, esum2, h2, acc2);

    finalize_kernel<<<blocks(NNODES * NCLS, TB), TB, 0, stream>>>(acc2, b2, out);
}
